// FC_CharNet_70832600646029
// MI455X (gfx1250) — compile-verified, matrix operand order checked by reading, not run on hardware
//
#include <hip/hip_runtime.h>
#include <math.h>

// ---------------------------------------------------------------------------
// CDNA5 (gfx1250, wave32) WMMA types: v_wmma_f32_16x16x32_bf16
// ---------------------------------------------------------------------------
typedef __attribute__((ext_vector_type(16))) __bf16           v16bf;
typedef __attribute__((ext_vector_type(16))) unsigned short   v16u;
typedef __attribute__((ext_vector_type(8)))  float            v8f;

union FragU { v16u u; v16bf b; uint4 q[2]; };

#define LDT 40   // LDS row stride in ushorts: 80B rows keep every chunk 16B-aligned

__device__ __forceinline__ unsigned short f2bf(float f) {
  union { float f; unsigned u; } x; x.f = f;
  unsigned r = x.u + 0x7fffu + ((x.u >> 16) & 1u);   // RNE (finite inputs)
  return (unsigned short)(r >> 16);
}

// A/B fragment from LDS tile (row-major [rows][LDT] bf16-as-ushort).
// ISA 16-bit 16x32 layout: lanes 0-15 row M=lane hold K 0..7 + 16..23;
// lanes 16-31 hold K 8..15 + 24..31.
__device__ __forceinline__ v16bf lds_frag(const unsigned short* t, int row0, int lane) {
  const uint4* p = (const uint4*)(t + (row0 + (lane & 15)) * LDT + ((lane >> 4) << 3));
  FragU f;
  f.q[0] = p[0];   // K+0..7
  f.q[1] = p[2];   // K+16..23 (+32 bytes)
  return f.b;
}

__device__ __forceinline__ v8f wmma_bf16(v16bf a, v16bf b, v8f c) {
  return __builtin_amdgcn_wmma_f32_16x16x32_bf16(false, a, false, b, (short)0, c, false, false);
}

// Guarded 32-byte fetch; validity/pointer are loop-invariant at the call site.
__device__ __forceinline__ void fetch32B(uint4 t[2], const unsigned short* p, bool v) {
  if (v) {
    const uint4* s = (const uint4*)p;
    t[0] = s[0]; t[1] = s[1];
  } else {
    t[0] = uint4{0, 0, 0, 0}; t[1] = uint4{0, 0, 0, 0};
  }
}

// ---------------------------------------------------------------------------
// WMMA GEMM:  Y[M,N] = act(X[M,K]_bf16 @ W[N,K]_bf16^T + bias)
// BM=BN=128, BK=32, 256 threads (8 waves); wave owns 64x32 -> 4x2 wmma tiles.
// K multiple of 32; ldx/ldw multiples of 16 (guaranteed by padding).
// Register prefetch + double-buffered LDS: one barrier per K-step, global
// latency and LDS stores hidden behind the WMMA burst.
// ---------------------------------------------------------------------------
__global__ __launch_bounds__(256) void wmma_gemm_kernel(
    const unsigned short* __restrict__ X, long ldx,
    const unsigned short* __restrict__ W, long ldw,
    const float* __restrict__ bias,
    float* __restrict__ Yf, long ldyf,
    unsigned short* __restrict__ Yb, long ldyb,
    int M, int N, int K, int relu)
{
  __shared__ unsigned short As[2][128 * LDT];
  __shared__ unsigned short Bs[2][128 * LDT];
  int tid  = threadIdx.x;
  int lane = tid & 31;
  int wave = tid >> 5;
  int wr = wave >> 2;   // 0..1 -> 64-row half
  int wc = wave & 3;    // 0..3 -> 32-col quarter
  int m0 = blockIdx.y * 128;
  int n0 = blockIdx.x * 128;

  // loop-invariant staging addresses / guards
  int r  = tid >> 1;
  int c0 = (tid & 1) << 4;
  bool va = (m0 + r) < M;
  bool vb = (n0 + r) < N;
  const unsigned short* xrow = X + (long)(va ? m0 + r : 0) * ldx + c0;
  const unsigned short* wrow = W + (long)(vb ? n0 + r : 0) * ldw + c0;
  unsigned short* const lofs = (unsigned short*)0 + (r * LDT + c0); // reuse as offset
  (void)lofs;

  v8f acc[4][2];
#pragma unroll
  for (int i = 0; i < 4; ++i)
#pragma unroll
    for (int j = 0; j < 2; ++j)
#pragma unroll
      for (int q = 0; q < 8; ++q) acc[i][j][q] = 0.f;

  uint4 ta[2], tb[2];
  fetch32B(ta, xrow, va);
  fetch32B(tb, wrow, vb);
  {
    uint4* da = (uint4*)(As[0] + r * LDT + c0);
    uint4* db = (uint4*)(Bs[0] + r * LDT + c0);
    da[0] = ta[0]; da[1] = ta[1];
    db[0] = tb[0]; db[1] = tb[1];
  }

  int cur = 0;
  for (int k0 = 0; k0 < K; k0 += 32) {
    bool more = (k0 + 32) < K;
    if (more) {                       // prefetch next slab into registers
      fetch32B(ta, xrow + k0 + 32, va);
      fetch32B(tb, wrow + k0 + 32, vb);
    }
    __syncthreads();                  // As[cur]/Bs[cur] ready; prev reads done
    v16bf bf0 = lds_frag(Bs[cur], wc * 32,      lane);
    v16bf bf1 = lds_frag(Bs[cur], wc * 32 + 16, lane);
#pragma unroll
    for (int i = 0; i < 4; ++i) {
      v16bf af = lds_frag(As[cur], wr * 64 + i * 16, lane);
      acc[i][0] = wmma_bf16(af, bf0, acc[i][0]);
      acc[i][1] = wmma_bf16(af, bf1, acc[i][1]);
    }
    if (more) {                       // stage into the other buffer
      uint4* da = (uint4*)(As[cur ^ 1] + r * LDT + c0);
      uint4* db = (uint4*)(Bs[cur ^ 1] + r * LDT + c0);
      da[0] = ta[0]; da[1] = ta[1];
      db[0] = tb[0]; db[1] = tb[1];
    }
    cur ^= 1;
  }

  int rbase = m0 + wr * 64 + ((lane >> 4) << 3);
  int cbase = n0 + wc * 32 + (lane & 15);
#pragma unroll
  for (int j = 0; j < 2; ++j) {
    int col = cbase + j * 16;
    if (col >= N) continue;
    float bv = bias ? bias[col] : 0.f;
#pragma unroll
    for (int i = 0; i < 4; ++i) {
#pragma unroll
      for (int q = 0; q < 8; ++q) {
        int row = rbase + i * 16 + q;
        if (row < M) {
          float v = acc[i][j][q] + bv;
          if (relu) v = v > 0.f ? v : 0.f;
          if (Yf) Yf[(long)row * ldyf + col] = v;
          if (Yb) Yb[(long)row * ldyb + col] = f2bf(v);
        }
      }
    }
  }
}

// ---------------------------------------------------------------------------
// Fused LSTM step: 32 state-rows x 32 h-cols per block; four gate accumulators
// over K=512; adds precomputed x-gates; cell update; bf16 h out (double buf),
// f32 c, hsum accumulation. Grid (16,10), 128 threads (4 waves).
// Register prefetch + double-buffered LDS (one barrier per K-step).
// ---------------------------------------------------------------------------
__global__ __launch_bounds__(128) void lstm_step_kernel(
    const unsigned short* __restrict__ h_in,   // [320,512] bf16
    unsigned short* __restrict__ h_out,        // [320,512] bf16
    float* __restrict__ c,                     // [320,512]
    float* __restrict__ hsum,                  // [320,512]
    const float* __restrict__ xg,              // [20480,2048] x@W_ih^T + (b_ih+b_hh)
    const unsigned short* __restrict__ Whh,    // [2048,512] bf16
    int s)
{
  __shared__ unsigned short As[2][32 * LDT];
  __shared__ unsigned short Bs[2][4][32 * LDT];
  int tid  = threadIdx.x;
  int lane = tid & 31;
  int wave = tid >> 5;
  int wr = wave >> 1, wc = wave & 1;
  int m0 = blockIdx.y * 32;
  int n0 = blockIdx.x * 32;
  int r  = tid >> 2;
  int c0 = (tid & 3) << 3;

  const unsigned short* hrow = h_in + (m0 + r) * 512 + c0;
  const unsigned short* wrow[4];
#pragma unroll
  for (int g = 0; g < 4; ++g) wrow[g] = Whh + (long)(g * 512 + n0 + r) * 512 + c0;

  v8f acc[4];
#pragma unroll
  for (int g = 0; g < 4; ++g)
#pragma unroll
    for (int q = 0; q < 8; ++q) acc[g][q] = 0.f;

  uint4 ta, tb[4];
  ta = *(const uint4*)hrow;
#pragma unroll
  for (int g = 0; g < 4; ++g) tb[g] = *(const uint4*)wrow[g];
  *(uint4*)(As[0] + r * LDT + c0) = ta;
#pragma unroll
  for (int g = 0; g < 4; ++g) *(uint4*)(Bs[0][g] + r * LDT + c0) = tb[g];

  int cur = 0;
  for (int k0 = 0; k0 < 512; k0 += 32) {
    bool more = (k0 + 32) < 512;
    if (more) {
      ta = *(const uint4*)(hrow + k0 + 32);
#pragma unroll
      for (int g = 0; g < 4; ++g) tb[g] = *(const uint4*)(wrow[g] + k0 + 32);
    }
    __syncthreads();
    v16bf af = lds_frag(As[cur], wr * 16, lane);
#pragma unroll
    for (int g = 0; g < 4; ++g) {
      v16bf bf = lds_frag(Bs[cur][g], wc * 16, lane);
      acc[g] = wmma_bf16(af, bf, acc[g]);
    }
    if (more) {
      *(uint4*)(As[cur ^ 1] + r * LDT + c0) = ta;
#pragma unroll
      for (int g = 0; g < 4; ++g) *(uint4*)(Bs[cur ^ 1][g] + r * LDT + c0) = tb[g];
    }
    cur ^= 1;
  }

  int rloc = wr * 16 + ((lane >> 4) << 3);
  int col  = n0 + wc * 16 + (lane & 15);
#pragma unroll
  for (int q = 0; q < 8; ++q) {
    int row = m0 + rloc + q;               // state row = p*32 + b
    int pp = row >> 5;
    int bb = row & 31;
    const float* xgp = xg + (((long)bb * 10 + pp) * 64 + s) * 2048;
    float iv = acc[0][q] + xgp[col];
    float fv = acc[1][q] + xgp[512 + col];
    float gv = acc[2][q] + xgp[1024 + col];
    float ov = acc[3][q] + xgp[1536 + col];
    float si = 1.f / (1.f + __expf(-iv));
    float sf = 1.f / (1.f + __expf(-fv));
    float so = 1.f / (1.f + __expf(-ov));
    float tg = tanhf(gv);
    long idx = (long)row * 512 + col;
    float cv = sf * c[idx] + si * tg;
    c[idx] = cv;
    float hv = so * tanhf(cv);
    h_out[idx] = f2bf(hv);
    hsum[idx] += hv;
  }
}

// ---------------------------------------------------------------------------
// Helper kernels
// ---------------------------------------------------------------------------
__global__ void pad_convert_kernel(const float* __restrict__ s, unsigned short* __restrict__ d,
                                   long total, int K, int Kp) {
  long i = (long)blockIdx.x * blockDim.x + threadIdx.x;
  if (i >= total) return;
  long row = i / Kp;
  int  k   = (int)(i - row * Kp);
  d[i] = (k < K) ? f2bf(s[row * K + k]) : (unsigned short)0;
}

__global__ void bias_sum_kernel(const float* a, const float* b, float* o, int n) {
  int i = blockIdx.x * blockDim.x + threadIdx.x;
  if (i < n) o[i] = a[i] + b[i];
}

__global__ void init_state_kernel(const float* hx0, const float* cx0,
                                  unsigned short* h0, float* c, float* hs, int n) {
  int i = blockIdx.x * blockDim.x + threadIdx.x;
  if (i < n) { h0[i] = f2bf(hx0[i]); c[i] = cx0[i]; hs[i] = 0.f; }
}

// cat rows [320, 544]: [q | other^T | me^T | zeros]
__global__ void build_cat_kernel(const float* hsum, const float* other,
                                 const float* me, unsigned short* cat) {
  int idx = blockIdx.x * blockDim.x + threadIdx.x;
  if (idx >= 320 * 544) return;
  int m = idx / 544, col = idx % 544;
  int p = m >> 5, b = m & 31;
  float v = 0.f;
  if (col < 512)       v = hsum[m * 512 + col];
  else if (col < 519)  v = other[(b * 10 + p) * 7 + (col - 512)];
  else if (col < 526)  v = me[(b * 10 + p) * 7 + (col - 519)];
  cat[idx] = f2bf(v);
}

__global__ void reduce_final_kernel(const float* e2, float* out) {
  int idx = blockIdx.x * blockDim.x + threadIdx.x;
  if (idx >= 32 * 256) return;
  float s = 0.f;
#pragma unroll
  for (int p = 0; p < 10; ++p) s += e2[p * 8192 + idx];
  out[idx] = s;
}

// ---------------------------------------------------------------------------
// Host launcher
// ---------------------------------------------------------------------------
extern "C" void kernel_launch(void* const* d_in, const int* in_sizes, int n_in,
                              void* d_out, int out_size, void* d_ws, size_t ws_size,
                              hipStream_t stream) {
  (void)in_sizes; (void)n_in; (void)out_size; (void)ws_size;
  const float* board   = (const float*)d_in[0];
  const float* order   = (const float*)d_in[1];
  const float* message = (const float*)d_in[2];
  const float* other   = (const float*)d_in[3];
  const float* me      = (const float*)d_in[4];
  const float* hx0     = (const float*)d_in[5];
  const float* cx0     = (const float*)d_in[6];
  const float* Wf[11] = { (const float*)d_in[7],  (const float*)d_in[9],  (const float*)d_in[11],
                          (const float*)d_in[13], (const float*)d_in[15], (const float*)d_in[17],
                          (const float*)d_in[19], (const float*)d_in[21], (const float*)d_in[22],
                          (const float*)d_in[25], (const float*)d_in[27] };
  const float* bb1 = (const float*)d_in[8],  *bb2 = (const float*)d_in[10], *bb3 = (const float*)d_in[12];
  const float* bo1 = (const float*)d_in[14], *bo2 = (const float*)d_in[16];
  const float* bm1 = (const float*)d_in[18], *bm2 = (const float*)d_in[20];
  const float* b_ih = (const float*)d_in[23], *b_hh = (const float*)d_in[24];
  const float* b1 = (const float*)d_in[26], *b2 = (const float*)d_in[28];
  float* out = (float*)d_out;

  const int R = 20480;   // B*P*S
  char* pws = (char*)d_ws;
  auto alloc = [&](size_t bytes) -> char* {
    char* r = pws; pws += (bytes + 255) & ~(size_t)255; return r;
  };
  typedef unsigned short u16;
  u16*  board_b = (u16*)alloc((size_t)R * 2848 * 2);
  u16*  order_b = (u16*)alloc((size_t)R * 64 * 2);
  u16*  msg_b   = (u16*)alloc((size_t)R * 64 * 2);
  u16*  h1_b    = (u16*)alloc((size_t)R * 1024 * 2);
  u16*  h2_b    = (u16*)alloc((size_t)R * 256 * 2);
  u16*  x_b     = (u16*)alloc((size_t)R * 512 * 2);
  u16*  o1_b    = (u16*)alloc((size_t)R * 128 * 2);
  u16*  m1_b    = (u16*)alloc((size_t)R * 64 * 2);
  float* xg     = (float*)alloc((size_t)R * 2048 * 4);
  const int wrows[11] = {1024, 256, 256, 128, 128,  64, 128, 2048, 2048, 256, 256};
  const int wK[11]    = {2835, 1024, 256, 40, 128,  40,  64,  512,  512, 526, 256};
  const int wKp[11]   = {2848, 1024, 256, 64, 128,  64,  64,  512,  512, 544, 256};
  u16* Wb[11];
  for (int i = 0; i < 11; ++i) Wb[i] = (u16*)alloc((size_t)wrows[i] * wKp[i] * 2);
  float* bsum  = (float*)alloc(2048 * 4);
  u16*   hA    = (u16*)alloc((size_t)320 * 512 * 2);
  u16*   hB    = (u16*)alloc((size_t)320 * 512 * 2);
  float* cst   = (float*)alloc((size_t)320 * 512 * 4);
  float* hsum  = (float*)alloc((size_t)320 * 512 * 4);
  u16*   cat_b = (u16*)alloc((size_t)320 * 544 * 2);
  u16*   e1_b  = (u16*)alloc((size_t)320 * 256 * 2);

  auto pad_convert = [&](const float* s, u16* d, long rows, int K, int Kp) {
    long total = rows * Kp;
    pad_convert_kernel<<<(int)((total + 255) / 256), 256, 0, stream>>>(s, d, total, K, Kp);
  };
  auto gemm = [&](const u16* X, long ldx, const u16* W, long ldw, const float* bias,
                  float* Yf, long ldyf, u16* Yb, long ldyb, int M, int N, int K, int relu) {
    dim3 grid((N + 127) / 128, (M + 127) / 128);
    wmma_gemm_kernel<<<grid, 256, 0, stream>>>(X, ldx, W, ldw, bias, Yf, ldyf, Yb, ldyb,
                                               M, N, K, relu);
  };

  // ---- convert inputs + weights to padded bf16 (board copy fits in L2) ----
  pad_convert(board,   board_b, R, 2835, 2848);
  pad_convert(order,   order_b, R,   40,   64);
  pad_convert(message, msg_b,   R,   40,   64);
  for (int i = 0; i < 11; ++i) pad_convert(Wf[i], Wb[i], wrows[i], wK[i], wKp[i]);
  bias_sum_kernel<<<(2048 + 255) / 256, 256, 0, stream>>>(b_ih, b_hh, bsum, 2048);
  init_state_kernel<<<(320 * 512 + 255) / 256, 256, 0, stream>>>(hx0, cx0, hA, cst, hsum, 320 * 512);

  // ---- branch MLPs; outputs concatenated into x_b[:, 0:512] ----
  gemm(board_b, 2848, Wb[0], 2848, bb1, nullptr, 0, h1_b,     1024, R, 1024, 2848, 1);
  gemm(h1_b,    1024, Wb[1], 1024, bb2, nullptr, 0, h2_b,      256, R,  256, 1024, 1);
  gemm(h2_b,     256, Wb[2],  256, bb3, nullptr, 0, x_b + 0,   512, R,  256,  256, 1);
  gemm(order_b,   64, Wb[3],   64, bo1, nullptr, 0, o1_b,      128, R,  128,   64, 1);
  gemm(o1_b,     128, Wb[4],  128, bo2, nullptr, 0, x_b + 256, 512, R,  128,  128, 1);
  gemm(msg_b,     64, Wb[5],   64, bm1, nullptr, 0, m1_b,       64, R,   64,   64, 1);
  gemm(m1_b,      64, Wb[6],   64, bm2, nullptr, 0, x_b + 384, 512, R,  128,   64, 1);

  // ---- batched x-gates, then 64 fused recurrent steps ----
  gemm(x_b, 512, Wb[7], 512, bsum, xg, 2048, nullptr, 0, R, 2048, 512, 0);
  for (int s = 0; s < 64; ++s) {
    const u16* hin = (s & 1) ? hB : hA;
    u16* hout      = (s & 1) ? hA : hB;
    lstm_step_kernel<<<dim3(16, 10), 128, 0, stream>>>(hin, hout, cst, hsum, xg, Wb[8], s);
  }

  // ---- head: cat -> 256 (ReLU) -> 256; e_char_total into d_out[8192:] ----
  build_cat_kernel<<<(320 * 544 + 255) / 256, 256, 0, stream>>>(hsum, other, me, cat_b);
  gemm(cat_b, 544, Wb[9], 544, b1, nullptr, 0, e1_b, 256, 320, 256, 544, 1);
  gemm(e1_b,  256, Wb[10], 256, b2, out + 8192, 256, nullptr, 0, 320, 256, 256, 0);
  reduce_final_kernel<<<(8192 + 255) / 256, 256, 0, stream>>>(out + 8192, out);
}
